// GCN_49151605735707
// MI455X (gfx1250) — matrix operand order repaired and checked by reading, not compiled, not run on hardware
//
#include <hip/hip_runtime.h>

typedef _Float16 half_t;
typedef __attribute__((ext_vector_type(16))) _Float16 v16h;
typedef __attribute__((ext_vector_type(8)))  _Float16 v8h;
typedef __attribute__((ext_vector_type(8)))  float    v8f;

#define N_NODES 50000
#define N_EDGES 800000
#define IN_F    128

// ---------------- degree / norm ----------------
__global__ void gcn_deg_kernel(const int* __restrict__ dst, float* __restrict__ deg, int e) {
    int i = blockIdx.x * blockDim.x + threadIdx.x;
    if (i < e) atomicAdd(&deg[dst[i]], 1.0f);
}

__global__ void gcn_norm_kernel(float* __restrict__ deg, int n) {
    int i = blockIdx.x * blockDim.x + threadIdx.x;
    if (i < n) {
        float d = deg[i];
        deg[i] = (d > 0.0f) ? rsqrtf(fmaxf(d, 1.0f)) : 0.0f;
    }
}

// ---------------- Xh = (x * norm) as f16 ----------------
// ncol = 128 -> row = idx >> 7
__global__ void gcn_prep_kernel(const float* __restrict__ x, const float* __restrict__ norm,
                                half_t* __restrict__ xh, unsigned total) {
    unsigned idx = blockIdx.x * blockDim.x + threadIdx.x;
    if (idx < total) xh[idx] = (half_t)(x[idx] * norm[idx >> 7]);
}

// ---------------- pack W (f32 row-major KxN) into WMMA B-operand tiles ----------------
// Tile (t = k0/32, c = col0/16) is 512 f16; lane l owns elements [l*16 .. l*16+15]:
//   element j = W[k0 + 8*(l>>4) + (j&7) + 16*(j>>3)][col0 + (l&15)]
// tc_shift = log2(ncol/16)
__global__ void gcn_packw_kernel(const float* __restrict__ W, half_t* __restrict__ Wp,
                                 int ncol, int tc_shift) {
    int idx = blockIdx.x * blockDim.x + threadIdx.x;
    int total = 128 * ncol;
    if (idx >= total) return;
    int tile   = idx >> 9;
    int within = idx & 511;
    int l = within >> 4;
    int j = within & 15;
    int t = tile >> tc_shift;
    int c = tile & ((1 << tc_shift) - 1);
    int srcK = t * 32 + 8 * (l >> 4) + (j & 7) + 16 * (j >> 3);
    int srcN = c * 16 + (l & 15);
    Wp[idx] = (half_t)W[srcK * ncol + srcN];
}

// ---------------- WMMA GEMM: Y[50000 x ncol] = Xh[50000 x 128] @ W[128 x ncol] ----------------
// grid.x = row tiles (16 rows each); one wave per 16-col tile (blockDim = 32 * ncol/16)
__global__ void gcn_gemm_wmma_kernel(const half_t* __restrict__ Xh,
                                     const half_t* __restrict__ Wp,
                                     float* __restrict__ Y, int ncol) {
    const int lane = threadIdx.x & 31;
    const int wave = threadIdx.x >> 5;
    const int row0 = blockIdx.x << 4;
    const int col0 = wave << 4;
    const int tiles_c = ncol >> 4;
    const int m  = lane & 15;
    const int hi = lane >> 4;

    const half_t* arow = Xh + (unsigned)(row0 + m) * IN_F;
    v8f acc = {};

#pragma unroll
    for (int t = 0; t < 4; ++t) {
        const int k0 = t * 32;
        // A fragment (16x32 f16): row-major 16B loads per ISA layout
        v8h alo = *(const v8h*)(arow + k0 + 8 * hi);        // K = k0 + 8*hi .. +7
        v8h ahi = *(const v8h*)(arow + k0 + 16 + 8 * hi);   // K = k0+16+8*hi .. +7
        v16h a = __builtin_shufflevector(alo, ahi,
                                         0, 1, 2, 3, 4, 5, 6, 7,
                                         8, 9, 10, 11, 12, 13, 14, 15);
        // B fragment: contiguous per-lane 32B from packed tile
        const half_t* bp = Wp + ((unsigned)(t * tiles_c + wave) << 9) + (lane << 4);
        v16h b = *(const v16h*)bp;

        acc = __builtin_amdgcn_wmma_f32_16x16x32_f16(false, a, false, b,
                                                     (short)0, acc, false, false);
    }

    // D/C layout: element (vgpr r, lane): M = r + 8*hi, N = lane&15
    // 32-bit element indexing (Y < 4GB) -> saddr + voffset stores
    unsigned base = (unsigned)(row0 + 8 * hi) * (unsigned)ncol + (unsigned)(col0 + m);
#pragma unroll
    for (int r = 0; r < 8; ++r) {
        Y[base + (unsigned)r * (unsigned)ncol] = acc[r];
    }
}

// ---------------- edge scatter-sum: agg[dst] += h[src], float4 chunks ----------------
// chunk_shift = log2(ncol/4); a wave covers one full feature row (coalesced)
__global__ void gcn_agg_kernel(const float* __restrict__ h, const int* __restrict__ src,
                               const int* __restrict__ dst, float* __restrict__ agg,
                               unsigned total, int chunk_shift, int ncol) {
    unsigned tid = blockIdx.x * blockDim.x + threadIdx.x;
    if (tid >= total) return;
    unsigned e = tid >> chunk_shift;
    unsigned c = tid & ((1u << chunk_shift) - 1u);
    int s = src[e];
    int d = dst[e];
    float4 v = ((const float4*)(h + (unsigned)s * (unsigned)ncol))[c];
    float* p = agg + (unsigned)d * (unsigned)ncol + c * 4u;
    atomicAdd(p + 0, v.x);
    atomicAdd(p + 1, v.y);
    atomicAdd(p + 2, v.z);
    atomicAdd(p + 3, v.w);
}

// ---------------- finalize: act = agg*norm + b (, relu); emit f16 next-input and/or f32 out ----
// f16 path also folds in the NEXT layer's leading h*norm multiply. col_shift = log2(ncol)
__global__ void gcn_finalize_kernel(const float* __restrict__ agg, const float* __restrict__ norm,
                                    const float* __restrict__ bias, half_t* xh_out, float* f_out,
                                    unsigned total, int col_shift, int do_relu) {
    unsigned idx = blockIdx.x * blockDim.x + threadIdx.x;
    if (idx >= total) return;
    unsigned row = idx >> col_shift;
    unsigned col = idx & ((1u << col_shift) - 1u);
    float nrm = norm[row];
    float v = agg[idx] * nrm + bias[col];
    if (do_relu) v = fmaxf(v, 0.0f);
    if (f_out)  f_out[idx] = v;
    if (xh_out) xh_out[idx] = (half_t)(v * nrm);
}

static inline int ilog2(int x) { int s = 0; while ((1 << s) < x) ++s; return s; }

extern "C" void kernel_launch(void* const* d_in, const int* in_sizes, int n_in,
                              void* d_out, int out_size, void* d_ws, size_t ws_size,
                              hipStream_t stream) {
    const float* features = (const float*)d_in[0];
    const int*   src      = (const int*)d_in[1];
    const int*   dst      = (const int*)d_in[2];
    const float* W1 = (const float*)d_in[3];
    const float* b1 = (const float*)d_in[4];
    const float* W2 = (const float*)d_in[5];
    const float* b2 = (const float*)d_in[6];
    const float* W3 = (const float*)d_in[7];
    const float* b3 = (const float*)d_in[8];
    float* out = (float*)d_out;

    const int N = N_NODES, E = N_EDGES, F = IN_F;

    // workspace carve-up (256B aligned)
    char* ws = (char*)d_ws;
    size_t off = 0;
    auto carve = [&](size_t bytes) -> void* {
        void* p = ws + off;
        off += (bytes + 255) & ~(size_t)255;
        return p;
    };
    float*  norm = (float*)carve((size_t)N * sizeof(float));          // deg -> norm in place
    half_t* Xh   = (half_t*)carve((size_t)N * F * sizeof(half_t));    // f16 GEMM input
    float*  hbuf = (float*)carve((size_t)N * F * sizeof(float));      // GEMM output
    float*  agg  = (float*)carve((size_t)N * F * sizeof(float));      // scatter accumulator
    half_t* Wp   = (half_t*)carve((size_t)F * F * sizeof(half_t));    // packed weights

    // degree + norm
    hipMemsetAsync(norm, 0, (size_t)N * sizeof(float), stream);
    gcn_deg_kernel<<<(E + 255) / 256, 256, 0, stream>>>(dst, norm, E);
    gcn_norm_kernel<<<(N + 255) / 256, 256, 0, stream>>>(norm, N);

    // layer-1 input: Xh = features * norm (f16)
    {
        unsigned total = (unsigned)N * F;
        gcn_prep_kernel<<<(total + 255) / 256, 256, 0, stream>>>(features, norm, Xh, total);
    }

    struct Layer { const float* W; const float* b; int ncol; };
    Layer layers[3] = { {W1, b1, 128}, {W2, b2, 128}, {W3, b3, 64} };

    for (int l = 0; l < 3; ++l) {
        const int ncol = layers[l].ncol;
        const int tiles_c = ncol / 16;
        const int col_shift = ilog2(ncol);

        // pack weights into WMMA B layout
        gcn_packw_kernel<<<(128 * ncol + 255) / 256, 256, 0, stream>>>(
            layers[l].W, Wp, ncol, ilog2(tiles_c));

        // WMMA GEMM: hbuf = Xh @ W
        gcn_gemm_wmma_kernel<<<N / 16, 32 * tiles_c, 0, stream>>>(Xh, Wp, hbuf, ncol);

        // scatter-sum over edges
        hipMemsetAsync(agg, 0, (size_t)N * ncol * sizeof(float), stream);
        {
            const int chunk_shift = col_shift - 2;           // ncol/4 float4 chunks
            unsigned total = (unsigned)E << chunk_shift;
            gcn_agg_kernel<<<(total + 255) / 256, 256, 0, stream>>>(
                hbuf, src, dst, agg, total, chunk_shift, ncol);
        }

        // finalize
        unsigned total = (unsigned)N * ncol;
        if (l < 2) {
            gcn_finalize_kernel<<<(total + 255) / 256, 256, 0, stream>>>(
                agg, norm, layers[l].b, Xh, (float*)nullptr, total, col_shift, 1);
        } else {
            gcn_finalize_kernel<<<(total + 255) / 256, 256, 0, stream>>>(
                agg, norm, layers[l].b, (half_t*)nullptr, out, total, col_shift, 0);
        }
    }
}
